// DeepShiftedInstructedAttentionBlock_72189810311258
// MI455X (gfx1250) — compile-verified, matrix operand order checked by reading, not run on hardware
//
#include <hip/hip_runtime.h>

typedef __bf16 bf16;
typedef __attribute__((ext_vector_type(16))) __bf16 v16bf;
typedef __attribute__((ext_vector_type(8)))  float  v8f;

namespace {
constexpr int Bb = 2, Cc = 192, Hh = 64, Ww = 64, Dd = 16, HEADS = 4;
constexpr int It = 16, Nn = 80, NP = 96;          // tokens: 16 instr + 64 content, padded to 96
constexpr int Pw = 1024;                          // 8*8*16 windows
constexpr int HS = 200;                           // h / o row stride (bf16 elems)
constexpr int QS = 64;                            // per-head q/k stride (dh 48 padded to 64)
constexpr int VS = 104;                           // vT / a row stride
constexpr int AS = 104;
constexpr int OFF_H  = 0;                         // bf16 [96][200] (38400B) unioned with f32 logits [96][96] (36864B)
constexpr int OFF_Q  = 38400;                     // bf16 [4][96][64] = 49152B
constexpr int OFF_K  = 87552;                     // bf16 [4][96][64] = 49152B
constexpr int OFF_VT = 136704;                    // bf16 [192][104]  = 39936B
constexpr int OFF_A  = 176640;                    // bf16 [96][104]   = 19968B
constexpr int OFF_O  = 196608;                    // bf16 [96][200]   = 38400B
constexpr int SMEM_BYTES = 235008;                // ~229.5 KB < 320 KB/WGP
constexpr float SCALE = 0.14433756729740643f;     // 1/sqrt(48)
}

__device__ __forceinline__ float wsum(float v) {
#pragma unroll
  for (int o = 16; o > 0; o >>= 1) v += __shfl_xor(v, o, 32);
  return v;
}
__device__ __forceinline__ float wmax(float v) {
#pragma unroll
  for (int o = 16; o > 0; o >>= 1) v = fmaxf(v, __shfl_xor(v, o, 32));
  return v;
}
__device__ __forceinline__ int ridx(int v) { return v < 56 ? 0 : (v < 60 ? 1 : 2); }

// A/B fragment for v_wmma_f32_16x16x32_bf16. Source is row-major [row][k] with
// given stride; for B this means the matrix is stored K-contiguous per column
// (i.e. transposed), which is how we lay out k, vT and the prepped weights.
__device__ __forceinline__ v16bf load_frag(const bf16* base, int row, int stride, int k0, int lane) {
  const bf16* p = base + row * stride + k0 + ((lane & 16) ? 8 : 0);
  v16bf f;
#pragma unroll
  for (int e = 0; e < 8; ++e) { f[e] = p[e]; f[8 + e] = p[16 + e]; }
  return f;
}

__device__ __forceinline__ v8f wmma_bf16(v16bf a, v16bf b, v8f c) {
  return __builtin_amdgcn_wmma_f32_16x16x32_bf16(false, a, false, b, (short)0, c, false, false);
}

// ---- prep: fp32 [K][N] weight -> bf16 [N][K] (K-contiguous for B fragments)
__global__ void wtconv_kernel(const float* __restrict__ W, bf16* __restrict__ out, int K, int N) {
  int i = blockIdx.x * blockDim.x + threadIdx.x;
  if (i >= K * N) return;
  int k = i / N, n = i - k * N;
  out[n * K + k] = (bf16)W[i];
}

// ---- prep: pos[b][h][m][n] for m,n in [0,80): rel-pos bias + instruction bias
__global__ void pos_kernel(const float* __restrict__ rel, const float* __restrict__ Wc,
                           const float* __restrict__ instr, const float* __restrict__ Wi,
                           const int* __restrict__ labels, float* __restrict__ pos) {
  int idx = blockIdx.x * blockDim.x + threadIdx.x;
  if (idx >= Bb * HEADS * Nn * Nn) return;
  int n = idx % Nn;
  int m = (idx / Nn) % Nn;
  int h = (idx / (Nn * Nn)) % HEADS;
  int b = idx / (Nn * Nn * HEADS);
  float v = 0.f;
  if (m >= It && n >= It) {
    int ti = m - It, tj = n - It;
    int r = ((ti >> 3) - (tj >> 3) + 7) * 15 + ((ti & 7) - (tj & 7) + 7);
    const float* rt = rel + r * 32;
#pragma unroll
    for (int k = 0; k < 32; ++k) v += rt[k] * Wc[k * HEADS + h];
  }
  if (m < It) {
    int lbl = labels[b * 2 + (m >> 3)];
    const float* e = instr + (size_t)(lbl * 8 + (m & 7)) * 32;
#pragma unroll
    for (int k = 0; k < 32; ++k) v += e[k] * Wi[k * HEADS + h];
  }
  if (n < It) {
    int lbl = labels[b * 2 + (n >> 3)];
    const float* e = instr + (size_t)(lbl * 8 + (n & 7)) * 32;
#pragma unroll
    for (int k = 0; k < 32; ++k) v += e[k] * Wi[k * HEADS + h];
  }
  pos[idx] = v;
}

// ---- main fused window-attention kernel: one workgroup per (window p, batch b)
__global__ void __launch_bounds__(256)
attn_window_kernel(const float* __restrict__ x_in, const float* __restrict__ xi,
                   const bf16* __restrict__ WqkvT, const float* __restrict__ bqkv,
                   const bf16* __restrict__ WoT, const float* __restrict__ bo,
                   const float* __restrict__ ln_g, const float* __restrict__ ln_b,
                   const float* __restrict__ pos, float* __restrict__ x_out,
                   int sh, int sw, int use_mask) {
  extern __shared__ char smem[];
  bf16*  hL = (bf16*)(smem + OFF_H);
  float* lg = (float*)(smem + OFF_H);   // logits reuse h's space after QKV
  bf16*  qL = (bf16*)(smem + OFF_Q);
  bf16*  kL = (bf16*)(smem + OFF_K);
  bf16*  vT = (bf16*)(smem + OFF_VT);
  bf16*  aL = (bf16*)(smem + OFF_A);
  bf16*  oL = (bf16*)(smem + OFF_O);

  const int wid  = threadIdx.x >> 5;
  const int lane = threadIdx.x & 31;
  const int b = blockIdx.y;
  const int p = blockIdx.x;
  const int ip1 = p >> 7, ip2 = (p >> 4) & 7, ip3 = p & 15;

  // ---------- stage 1: gather x_all rows + LayerNorm -> h bf16 ----------
  for (int m = wid; m < NP; m += 8) {
    float vals[6];
    if (m >= Nn) {
#pragma unroll
      for (int j = 0; j < 6; ++j) vals[j] = 0.f;
    } else if (m < It) {
#pragma unroll
      for (int j = 0; j < 6; ++j) vals[j] = xi[((size_t)b * It + m) * Cc + lane + 32 * j];
    } else {
      int t = m - It;
      int hh = (((t >> 3) << 3) + ip1 + sh) & 63;   // strided window + shift (roll folded in)
      int wc = (((t & 7) << 3) + ip2 + sw) & 63;
#pragma unroll
      for (int j = 0; j < 6; ++j) {
        int c = lane + 32 * j;
        vals[j] = x_in[((((size_t)b * Cc + c) * Hh + hh) * Ww + wc) * Dd + ip3];
      }
    }
    float s = 0.f;
#pragma unroll
    for (int j = 0; j < 6; ++j) s += vals[j];
    float mu = wsum(s) * (1.f / 192.f);
    float v2 = 0.f;
#pragma unroll
    for (int j = 0; j < 6; ++j) { float d = vals[j] - mu; v2 += d * d; }
    float inv = rsqrtf(wsum(v2) * (1.f / 192.f) + 1e-5f);
#pragma unroll
    for (int j = 0; j < 6; ++j) {
      int c = lane + 32 * j;
      float hv = (m < Nn) ? (vals[j] - mu) * inv * ln_g[c] + ln_b[c] : 0.f;
      hL[m * HS + c] = (bf16)hv;
    }
  }
  __syncthreads();

  // ---------- stage 2: QKV = h @ Wqkv + b  (WMMA, 6x36 tiles, K=192) ----------
  for (int i = threadIdx.x; i < HEADS * NP * 16; i += 256) {  // zero dh-pad cols 48..63
    int head = i / (NP * 16), rem = i % (NP * 16);
    int row = rem >> 4, col = 48 + (rem & 15);
    qL[(head * NP + row) * QS + col] = (bf16)0.f;
    kL[(head * NP + row) * QS + col] = (bf16)0.f;
  }
  for (int t = wid; t < 216; t += 8) {
    int mt = t / 36, nt = t % 36;
    int arow = mt * 16 + (lane & 15);
    int brow = nt * 16 + (lane & 15);
    v8f acc = {};
#pragma unroll
    for (int kc = 0; kc < 6; ++kc)
      acc = wmma_bf16(load_frag(hL, arow, HS, kc * 32, lane),
                      load_frag(WqkvT, brow, 192, kc * 32, lane), acc);
    int n = brow;
    float bias = bqkv[n];
#pragma unroll
    for (int r = 0; r < 8; ++r) {
      int m = mt * 16 + r + ((lane & 16) ? 8 : 0);
      float val = acc[r] + bias;
      if (n < 192) {
        int head = n / 48, col = n - head * 48;
        qL[(head * NP + m) * QS + col] = (bf16)val;
      } else if (n < 384) {
        int c2 = n - 192, head = c2 / 48, col = c2 - head * 48;
        kL[(head * NP + m) * QS + col] = (bf16)val;
      } else {
        int c2 = n - 384;                          // v stored transposed [C][key]
        vT[c2 * VS + m] = (bf16)((m < Nn) ? val : 0.f);  // zero pad keys 80..95
      }
    }
  }
  __syncthreads();

  // ---------- stage 3: per-head attention (only content query rows 16..79) ----------
  for (int head = 0; head < HEADS; ++head) {
    const bf16* qh = qL + head * NP * QS;
    const bf16* kh = kL + head * NP * QS;
    // logits = q @ k^T * scale + pos (+ shift mask), 4x6 tiles, K=64
    for (int t = wid; t < 24; t += 8) {
      int mt = 1 + t / 6, nt = t % 6;
      v8f acc = {};
#pragma unroll
      for (int kc = 0; kc < 2; ++kc)
        acc = wmma_bf16(load_frag(qh, mt * 16 + (lane & 15), QS, kc * 32, lane),
                        load_frag(kh, nt * 16 + (lane & 15), QS, kc * 32, lane), acc);
      int n = nt * 16 + (lane & 15);
#pragma unroll
      for (int r = 0; r < 8; ++r) {
        int m = mt * 16 + r + ((lane & 16) ? 8 : 0);
        float val;
        if (n >= Nn) {
          val = -1e9f;                              // padded keys
        } else {
          val = acc[r] * SCALE + pos[(((size_t)b * HEADS + head) * Nn + m) * Nn + n];
          if (use_mask) {
            bool masked;
            if (n < It) {
              masked = true;                        // content queries masked from instr keys
            } else {
              int ti = m - It, tj = n - It;
              int ri = ridx(((ti >> 3) << 3) + ip1) * 3 + ridx(((ti & 7) << 3) + ip2);
              int rj = ridx(((tj >> 3) << 3) + ip1) * 3 + ridx(((tj & 7) << 3) + ip2);
              masked = (ri != rj);
            }
            if (masked) val -= 1e9f;
          }
        }
        lg[m * NP + n] = val;
      }
    }
    __syncthreads();
    // row softmax (wave-per-row over 96 keys)
    for (int m = It + wid; m < Nn; m += 8) {
      float x0 = lg[m * NP + lane], x1 = lg[m * NP + lane + 32], x2 = lg[m * NP + lane + 64];
      float mx = wmax(fmaxf(x0, fmaxf(x1, x2)));
      float e0 = __expf(x0 - mx), e1 = __expf(x1 - mx), e2 = __expf(x2 - mx);
      float inv = 1.f / wsum(e0 + e1 + e2);
      aL[m * AS + lane]      = (bf16)(e0 * inv);
      aL[m * AS + lane + 32] = (bf16)(e1 * inv);
      aL[m * AS + lane + 64] = (bf16)(e2 * inv);
    }
    __syncthreads();
    // o_head = a @ v_head, 4x3 tiles, K=96
    for (int t = wid; t < 12; t += 8) {
      int mt = 1 + t / 3, nt = t % 3;
      v8f acc = {};
#pragma unroll
      for (int kc = 0; kc < 3; ++kc)
        acc = wmma_bf16(load_frag(aL, mt * 16 + (lane & 15), AS, kc * 32, lane),
                        load_frag(vT, head * 48 + nt * 16 + (lane & 15), VS, kc * 32, lane), acc);
      int n = head * 48 + nt * 16 + (lane & 15);
#pragma unroll
      for (int r = 0; r < 8; ++r) {
        int m = mt * 16 + r + ((lane & 16) ? 8 : 0);
        oL[m * HS + n] = (bf16)acc[r];
      }
    }
    __syncthreads();
  }

  // ---------- stage 4: out = x + o @ Wo + bo, content rows only, 4x12 tiles ----------
  for (int t = wid; t < 48; t += 8) {
    int mt = 1 + t / 12, nt = t % 12;
    v8f acc = {};
#pragma unroll
    for (int kc = 0; kc < 6; ++kc)
      acc = wmma_bf16(load_frag(oL, mt * 16 + (lane & 15), HS, kc * 32, lane),
                      load_frag(WoT, nt * 16 + (lane & 15), 192, kc * 32, lane), acc);
    int n = nt * 16 + (lane & 15);
    float bias = bo[n];
#pragma unroll
    for (int r = 0; r < 8; ++r) {
      int m = mt * 16 + r + ((lane & 16) ? 8 : 0);
      int tk = m - It;
      int hh = (((tk >> 3) << 3) + ip1 + sh) & 63;
      int wc = (((tk & 7) << 3) + ip2 + sw) & 63;
      size_t gi = ((((size_t)b * Cc + n) * Hh + hh) * Ww + wc) * Dd + ip3;
      x_out[gi] = x_in[gi] + acc[r] + bias;
    }
  }
}

extern "C" void kernel_launch(void* const* d_in, const int* in_sizes, int n_in,
                              void* d_out, int out_size, void* d_ws, size_t ws_size,
                              hipStream_t stream) {
  (void)in_sizes; (void)n_in; (void)out_size; (void)ws_size;
  const float* x     = (const float*)d_in[0];
  const float* xi0   = (const float*)d_in[1];
  const float* xi1   = (const float*)d_in[2];
  const int*   lbl   = (const int*)d_in[3];
  const float* ln_g0 = (const float*)d_in[4];
  const float* ln_b0 = (const float*)d_in[5];
  const float* Wqkv0 = (const float*)d_in[6];
  const float* bqkv0 = (const float*)d_in[7];
  const float* Wo0   = (const float*)d_in[8];
  const float* bo0   = (const float*)d_in[9];
  const float* rel0  = (const float*)d_in[10];
  const float* Wc0   = (const float*)d_in[11];
  const float* ins0  = (const float*)d_in[12];
  const float* Wi0   = (const float*)d_in[13];
  const float* ln_g1 = (const float*)d_in[14];
  const float* ln_b1 = (const float*)d_in[15];
  const float* Wqkv1 = (const float*)d_in[16];
  const float* bqkv1 = (const float*)d_in[17];
  const float* Wo1   = (const float*)d_in[18];
  const float* bo1   = (const float*)d_in[19];
  const float* rel1  = (const float*)d_in[20];
  const float* Wc1   = (const float*)d_in[21];
  const float* ins1  = (const float*)d_in[22];
  const float* Wi1   = (const float*)d_in[23];

  char* ws = (char*)d_ws;
  size_t off = 0;
  auto take = [&](size_t bytes) { char* q = ws + off; off = (off + bytes + 255) & ~(size_t)255; return q; };
  float* x_mid  = (float*)take((size_t)Bb * Cc * Hh * Ww * Dd * sizeof(float));
  bf16* WqkvT0  = (bf16*)take((size_t)576 * 192 * sizeof(bf16));
  bf16* WqkvT1  = (bf16*)take((size_t)576 * 192 * sizeof(bf16));
  bf16* WoT0    = (bf16*)take((size_t)192 * 192 * sizeof(bf16));
  bf16* WoT1    = (bf16*)take((size_t)192 * 192 * sizeof(bf16));
  float* pos0   = (float*)take((size_t)Bb * HEADS * Nn * Nn * sizeof(float));
  float* pos1   = (float*)take((size_t)Bb * HEADS * Nn * Nn * sizeof(float));

  hipFuncSetAttribute((const void*)attn_window_kernel,
                      hipFuncAttributeMaxDynamicSharedMemorySize, SMEM_BYTES);

  // prep: weight transpose+bf16, pos bias tables
  {
    int nq = 192 * 576, no = 192 * 192, np = Bb * HEADS * Nn * Nn;
    wtconv_kernel<<<(nq + 255) / 256, 256, 0, stream>>>(Wqkv0, WqkvT0, 192, 576);
    wtconv_kernel<<<(nq + 255) / 256, 256, 0, stream>>>(Wqkv1, WqkvT1, 192, 576);
    wtconv_kernel<<<(no + 255) / 256, 256, 0, stream>>>(Wo0, WoT0, 192, 192);
    wtconv_kernel<<<(no + 255) / 256, 256, 0, stream>>>(Wo1, WoT1, 192, 192);
    pos_kernel<<<(np + 255) / 256, 256, 0, stream>>>(rel0, Wc0, ins0, Wi0, lbl, pos0);
    pos_kernel<<<(np + 255) / 256, 256, 0, stream>>>(rel1, Wc1, ins1, Wi1, lbl, pos1);
  }

  dim3 grid(Pw, Bb);
  // block 0: no shift, no mask -> x_mid
  attn_window_kernel<<<grid, 256, SMEM_BYTES, stream>>>(
      x, xi0, WqkvT0, bqkv0, WoT0, bo0, ln_g0, ln_b0, pos0, x_mid, 0, 0, 0);
  // block 1: shift (4,4,0) folded into indexing, analytic mask -> d_out
  attn_window_kernel<<<grid, 256, SMEM_BYTES, stream>>>(
      x_mid, xi1, WqkvT1, bqkv1, WoT1, bo1, ln_g1, ln_b1, pos1, (float*)d_out, 4, 4, 1);
}
